// SwigluMoEBlock_12266426597724
// MI455X (gfx1250) — matrix operand order repaired and checked by reading, not compiled, not run on hardware
//
#include <hip/hip_runtime.h>
#include <hip/hip_bf16.h>
#include <math.h>

// ---------------------------------------------------------------------------
// SwiGLU MoE (T=4096, H=2048, I=2048, E=8, top-2) for gfx1250 (MI455X).
// bf16 WMMA grouped GEMMs over routed tokens. Weights + hidden pre-converted
// to bf16 once when workspace allows (no cvt in hot loop, half the bytes).
// Each wave: 32M x 64N strip = 2 M-tiles x 4 N-tiles (8 accumulators),
// per K-step: 4 A loads + 8 B loads -> 8 WMMAs (1.5 b128 loads per WMMA).
// Block: 8 waves = 2 M-groups x 4 N-groups -> 64 x 256 tile.
// ---------------------------------------------------------------------------

typedef __attribute__((ext_vector_type(16))) __bf16 v16bf;
typedef __attribute__((ext_vector_type(8)))  __bf16 v8bf;
typedef __attribute__((ext_vector_type(8)))  float  v8f;

#define T_TOK 4096
#define H_DIM 2048
#define I_DIM 2048
#define TWO_I 4096
#define E_NUM 8

__device__ __forceinline__ float fast_sigmoid(float x) {
    return __builtin_amdgcn_rcpf(1.0f + __expf(-x));   // v_rcp_f32, no div chain
}

__device__ __forceinline__ void put4(v16bf& v, int base, float4 f) {
    v[base + 0] = (__bf16)f.x;
    v[base + 1] = (__bf16)f.y;
    v[base + 2] = (__bf16)f.z;
    v[base + 3] = (__bf16)f.w;
}

__device__ __forceinline__ v16bf join16(v8bf r0, v8bf r1) {
    return __builtin_shufflevector(r0, r1, 0, 1, 2, 3, 4, 5, 6, 7,
                                   8, 9, 10, 11, 12, 13, 14, 15);
}

// A fragment (16x32 bf16): lane m=L%16, half=L/16; K runs {k0+8h..+7, k0+16+8h..+7}
__device__ __forceinline__ v16bf load_a_bf16(const __bf16* row, int k0, int half) {
    v8bf r0 = *(const v8bf*)(row + k0 + 8 * half);
    v8bf r1 = *(const v8bf*)(row + k0 + 16 + 8 * half);
    return join16(r0, r1);
}
__device__ __forceinline__ v16bf load_a_f32(const float* row, int k0, int half) {
    const float4* p0 = (const float4*)(row + k0 + 8 * half);
    const float4* p1 = (const float4*)(row + k0 + 16 + 8 * half);
    v16bf a;
    put4(a, 0, p0[0]); put4(a, 4, p0[1]); put4(a, 8, p1[0]); put4(a, 12, p1[1]);
    return a;
}
// B fragment (32x16 bf16): lane n=L%16, half=L/16; K runs {k0+8h..+7, k0+16+8h..+7}
__device__ __forceinline__ v16bf load_b_bf16(const __bf16* row, int k0, int half) {
    v8bf r0 = *(const v8bf*)(row + k0 + 8 * half);
    v8bf r1 = *(const v8bf*)(row + k0 + 16 + 8 * half);
    return join16(r0, r1);
}
__device__ __forceinline__ v16bf load_b_f32(const float* row, int k0, int half) {
    const float4* p0 = (const float4*)(row + k0 + 8 * half);
    const float4* p1 = (const float4*)(row + k0 + 16 + 8 * half);
    v16bf b;
    put4(b, 0, p0[0]); put4(b, 4, p0[1]); put4(b, 8, p1[0]); put4(b, 12, p1[1]);
    return b;
}

// ---------------------------------------------------------------------------
// fp32 -> bf16 bulk conversion (8 elements per thread, 16B stores)
// ---------------------------------------------------------------------------
__global__ __launch_bounds__(256) void cvt_bf16_kernel(
    const float* __restrict__ src, __bf16* __restrict__ dst, size_t n8)
{
    size_t i = (size_t)blockIdx.x * blockDim.x + threadIdx.x;
    if (i >= n8) return;
    const float4* s = (const float4*)src + 2 * i;
    float4 f0 = s[0], f1 = s[1];
    v8bf v;
    v[0] = (__bf16)f0.x; v[1] = (__bf16)f0.y; v[2] = (__bf16)f0.z; v[3] = (__bf16)f0.w;
    v[4] = (__bf16)f1.x; v[5] = (__bf16)f1.y; v[6] = (__bf16)f1.z; v[7] = (__bf16)f1.w;
    *((v8bf*)dst + i) = v;
}

// ---------------------------------------------------------------------------
// Router: one wave per token; top-2; gate = 2-way softmax of top-2 logits.
// ---------------------------------------------------------------------------
__global__ __launch_bounds__(256) void router_kernel(
    const float* __restrict__ hs, const float* __restrict__ rw,
    int* __restrict__ counts, int* __restrict__ top_idx, float* __restrict__ top_w)
{
    int wid  = threadIdx.x >> 5;
    int lane = threadIdx.x & 31;
    int t = blockIdx.x * 8 + wid;
    if (t >= T_TOK) return;

    const float* hrow = hs + (size_t)t * H_DIM;
    float acc[E_NUM];
#pragma unroll
    for (int e = 0; e < E_NUM; ++e) acc[e] = 0.0f;

    for (int h = lane; h < H_DIM; h += 32) {
        float x = hrow[h];
#pragma unroll
        for (int e = 0; e < E_NUM; ++e) acc[e] += x * rw[e * H_DIM + h];
    }
#pragma unroll
    for (int e = 0; e < E_NUM; ++e) {
#pragma unroll
        for (int off = 16; off > 0; off >>= 1)
            acc[e] += __shfl_xor(acc[e], off, 32);
    }
    if (lane == 0) {
        int a = 0;
#pragma unroll
        for (int e = 1; e < E_NUM; ++e) if (acc[e] > acc[a]) a = e;
        int b = (a == 0) ? 1 : 0;
#pragma unroll
        for (int e = 0; e < E_NUM; ++e) if (e != a && acc[e] > acc[b]) b = e;
        float w0 = __builtin_amdgcn_rcpf(1.0f + __expf(acc[b] - acc[a]));
        top_idx[2 * t + 0] = a;
        top_idx[2 * t + 1] = b;
        top_w[2 * t + 0] = w0;
        top_w[2 * t + 1] = 1.0f - w0;
        atomicAdd(&counts[a], 1);
        atomicAdd(&counts[b], 1);
    }
}

__global__ void scan_kernel(const int* __restrict__ counts, int* __restrict__ offsets)
{
    if (threadIdx.x == 0 && blockIdx.x == 0) {
        int s = 0;
        for (int e = 0; e < E_NUM; ++e) { offsets[e] = s; s += counts[e]; }
        offsets[E_NUM] = s;
    }
}

__global__ __launch_bounds__(256) void scatter_kernel(
    const int* __restrict__ top_idx, const float* __restrict__ top_w,
    const int* __restrict__ offsets, int* __restrict__ counts2,
    int* __restrict__ rows, float* __restrict__ rowgate)
{
    int t = blockIdx.x * blockDim.x + threadIdx.x;
    if (t >= T_TOK) return;
#pragma unroll
    for (int k = 0; k < 2; ++k) {
        int e = top_idx[2 * t + k];
        int p = atomicAdd(&counts2[e], 1);
        int idx = offsets[e] + p;
        rows[idx] = t;
        rowgate[idx] = top_w[2 * t + k];
    }
}

// ---------------------------------------------------------------------------
// fc1 + interleaved SwiGLU.
// ---------------------------------------------------------------------------
template <int PRE>
__global__ __launch_bounds__(256) void fc1_swiglu_kernel(
    const float* __restrict__ hs,   const __bf16* __restrict__ hsb,
    const float* __restrict__ fc1w, const __bf16* __restrict__ fc1wb,
    const float* __restrict__ fc1b,
    const int* __restrict__ counts, const int* __restrict__ offsets,
    const int* __restrict__ rows, __bf16* __restrict__ y)
{
    int e   = blockIdx.z;
    int cnt = counts[e];
    int mbase = blockIdx.y * 64;
    if (mbase >= cnt) return;
    int seg = offsets[e];

    int wid  = threadIdx.x >> 5;
    int lane = threadIdx.x & 31;
    int wm = wid >> 2;            // 0..1  (M group)
    int wn = wid & 3;             // 0..3  (N group)
    int n    = lane & 15;
    int half = lane >> 4;

    int j0      = blockIdx.x * 256 + wn * 64;   // 4 col tiles: j0 + 16*nt
    int m_tile0 = mbase + wm * 32;              // 2 row tiles: m_tile0 + 16*mt

    // A source rows (lane holds M = lane%16), clamped for padding
    const float*  arowf[2];
    const __bf16* arowb[2];
#pragma unroll
    for (int mt = 0; mt < 2; ++mt) {
        int am  = m_tile0 + mt * 16 + n;
        int amc = (am < cnt) ? am : 0;
        int tokA = rows[seg + amc];
        arowf[mt] = hs  + (size_t)tokA * H_DIM;
        arowb[mt] = hsb + (size_t)tokA * H_DIM;
    }

    const float*  browf[4];
    const __bf16* browb[4];
#pragma unroll
    for (int nt = 0; nt < 4; ++nt) {
        size_t r = (size_t)e * TWO_I + (size_t)(j0 + nt * 16 + n);
        browf[nt] = fc1w  + r * H_DIM;
        browb[nt] = fc1wb + r * H_DIM;
    }

    v8f c[2][4] = {};
    for (int k0 = 0; k0 < H_DIM; k0 += 32) {
        v16bf a0, a1;
        if constexpr (PRE) {
            a0 = load_a_bf16(arowb[0], k0, half);
            a1 = load_a_bf16(arowb[1], k0, half);
        } else {
            a0 = load_a_f32(arowf[0], k0, half);
            a1 = load_a_f32(arowf[1], k0, half);
        }
#pragma unroll
        for (int nt = 0; nt < 4; ++nt) {
            v16bf b;
            if constexpr (PRE) b = load_b_bf16(browb[nt], k0, half);
            else               b = load_b_f32(browf[nt], k0, half);
            c[0][nt] = __builtin_amdgcn_wmma_f32_16x16x32_bf16(
                           false, a0, false, b, (short)0, c[0][nt], false, false);
            c[1][nt] = __builtin_amdgcn_wmma_f32_16x16x32_bf16(
                           false, a1, false, b, (short)0, c[1][nt], false, false);
        }
    }

#pragma unroll
    for (int nt = 0; nt < 4; ++nt) {
        int j = j0 + nt * 16 + n;
        float bias = fc1b[(size_t)e * TWO_I + (size_t)j];
#pragma unroll
        for (int mt = 0; mt < 2; ++mt) {
            int m_tile = m_tile0 + mt * 16;
#pragma unroll
            for (int r = 0; r < 8; ++r) {
                float v = c[mt][nt][r] + bias;           // col j, row m_tile+r+8h
                float other = __shfl_xor(v, 1, 32);      // partner col j^1
                int mrow = m_tile + r + 8 * half;
                if (((n & 1) == 0) && (mrow < cnt)) {
                    float xg = fminf(v, 7.0f);
                    float xl = fminf(fmaxf(other, -7.0f), 7.0f);
                    float yv = xg * fast_sigmoid(1.702f * xg) * (xl + 1.0f);
                    y[(size_t)(seg + mrow) * I_DIM +
                      (size_t)(((j0 + nt * 16) >> 1) + (n >> 1))] = (__bf16)yv;
                }
            }
        }
    }
}

// ---------------------------------------------------------------------------
// fc2 + gated combine. Exactly 2 float atomics per out element (bitwise
// deterministic since fp add of two values is commutative).
// ---------------------------------------------------------------------------
template <int PRE>
__global__ __launch_bounds__(256) void fc2_combine_kernel(
    const __bf16* __restrict__ y,
    const float* __restrict__ fc2w, const __bf16* __restrict__ fc2wb,
    const float* __restrict__ fc2b,
    const int* __restrict__ counts, const int* __restrict__ offsets,
    const int* __restrict__ rows, const float* __restrict__ rowgate,
    float* __restrict__ out)
{
    int e   = blockIdx.z;
    int cnt = counts[e];
    int mbase = blockIdx.y * 64;
    if (mbase >= cnt) return;
    int seg = offsets[e];

    int wid  = threadIdx.x >> 5;
    int lane = threadIdx.x & 31;
    int wm = wid >> 2;
    int wn = wid & 3;
    int n    = lane & 15;
    int half = lane >> 4;

    int h0      = blockIdx.x * 256 + wn * 64;
    int m_tile0 = mbase + wm * 32;

    const __bf16* yrow[2];
#pragma unroll
    for (int mt = 0; mt < 2; ++mt) {
        int am  = m_tile0 + mt * 16 + n;
        int amc = (am < cnt) ? am : 0;
        yrow[mt] = y + (size_t)(seg + amc) * I_DIM;
    }

    const float*  browf[4];
    const __bf16* browb[4];
#pragma unroll
    for (int nt = 0; nt < 4; ++nt) {
        size_t r = (size_t)e * H_DIM + (size_t)(h0 + nt * 16 + n);
        browf[nt] = fc2w  + r * I_DIM;
        browb[nt] = fc2wb + r * I_DIM;
    }

    v8f c[2][4] = {};
    for (int k0 = 0; k0 < I_DIM; k0 += 32) {
        v16bf a0 = load_a_bf16(yrow[0], k0, half);
        v16bf a1 = load_a_bf16(yrow[1], k0, half);
#pragma unroll
        for (int nt = 0; nt < 4; ++nt) {
            v16bf b;
            if constexpr (PRE) b = load_b_bf16(browb[nt], k0, half);
            else               b = load_b_f32(browf[nt], k0, half);
            c[0][nt] = __builtin_amdgcn_wmma_f32_16x16x32_bf16(
                           false, a0, false, b, (short)0, c[0][nt], false, false);
            c[1][nt] = __builtin_amdgcn_wmma_f32_16x16x32_bf16(
                           false, a1, false, b, (short)0, c[1][nt], false, false);
        }
    }

    // hoist per-row gate / token
    float gv[2][8]; int tok[2][8]; bool val[2][8];
#pragma unroll
    for (int mt = 0; mt < 2; ++mt) {
#pragma unroll
        for (int r = 0; r < 8; ++r) {
            int mrow = m_tile0 + mt * 16 + r + 8 * half;
            bool v = mrow < cnt;
            int idx = seg + (v ? mrow : 0);
            gv[mt][r]  = rowgate[idx];
            tok[mt][r] = rows[idx];
            val[mt][r] = v;
        }
    }
#pragma unroll
    for (int nt = 0; nt < 4; ++nt) {
        int h = h0 + nt * 16 + n;
        float bias = fc2b[(size_t)e * H_DIM + (size_t)h];
#pragma unroll
        for (int mt = 0; mt < 2; ++mt) {
#pragma unroll
            for (int r = 0; r < 8; ++r) {
                if (val[mt][r]) {
                    atomicAdd(out + (size_t)tok[mt][r] * H_DIM + (size_t)h,
                              gv[mt][r] * (c[mt][nt][r] + bias));
                }
            }
        }
    }
}

// ---------------------------------------------------------------------------
// Launch
// ---------------------------------------------------------------------------
extern "C" void kernel_launch(void* const* d_in, const int* in_sizes, int n_in,
                              void* d_out, int out_size, void* d_ws, size_t ws_size,
                              hipStream_t stream)
{
    const float* hs   = (const float*)d_in[0];   // [T, H]
    const float* rw   = (const float*)d_in[1];   // [E, H]
    const float* fc1w = (const float*)d_in[2];   // [E, 2I, H]
    const float* fc1b = (const float*)d_in[3];   // [E, 2I]
    const float* fc2w = (const float*)d_in[4];   // [E, H, I]
    const float* fc2b = (const float*)d_in[5];   // [E, H]
    float* out = (float*)d_out;                  // [T, H]

    char* ws = (char*)d_ws;
    size_t off = 0;
    int*    counts  = (int*)(ws + off); off += 32;
    int*    counts2 = (int*)(ws + off); off += 32;
    int*    offsets = (int*)(ws + off); off += 64;          // 9 ints, padded
    int*    top_idx = (int*)(ws + off); off += 8  * T_TOK;  // 2T ints
    float*  top_w   = (float*)(ws + off); off += 8 * T_TOK;
    int*    rows    = (int*)(ws + off); off += 8 * T_TOK;
    float*  rowgate = (float*)(ws + off); off += 8 * T_TOK;
    __bf16* ybuf    = (__bf16*)(ws + off); off += (size_t)2 * T_TOK * I_DIM * 2;

    size_t hs_n   = (size_t)T_TOK * H_DIM;
    size_t fc1w_n = (size_t)E_NUM * TWO_I * H_DIM;
    size_t fc2w_n = (size_t)E_NUM * H_DIM * I_DIM;

    __bf16* hsb    = (__bf16*)(ws + off); size_t off_pre = off + hs_n * 2;
    __bf16* fc1wb  = (__bf16*)(ws + off_pre); off_pre += fc1w_n * 2;
    __bf16* fc2wb  = (__bf16*)(ws + off_pre); off_pre += fc2w_n * 2;

    bool pre = (ws_size >= off_pre);

    hipMemsetAsync(d_out, 0, (size_t)T_TOK * H_DIM * sizeof(float), stream);
    hipMemsetAsync(ws, 0, 128, stream);

    if (pre) {
        cvt_bf16_kernel<<<(unsigned)((hs_n / 8 + 255) / 256), 256, 0, stream>>>(
            hs, hsb, hs_n / 8);
        cvt_bf16_kernel<<<(unsigned)((fc1w_n / 8 + 255) / 256), 256, 0, stream>>>(
            fc1w, fc1wb, fc1w_n / 8);
        cvt_bf16_kernel<<<(unsigned)((fc2w_n / 8 + 255) / 256), 256, 0, stream>>>(
            fc2w, fc2wb, fc2w_n / 8);
    }

    router_kernel<<<T_TOK / 8, 256, 0, stream>>>(hs, rw, counts, top_idx, top_w);
    scan_kernel<<<1, 32, 0, stream>>>(counts, offsets);
    scatter_kernel<<<T_TOK / 256, 256, 0, stream>>>(top_idx, top_w, offsets,
                                                    counts2, rows, rowgate);

    dim3 g1(TWO_I / 256, T_TOK / 64, E_NUM);
    dim3 g2(H_DIM / 256, T_TOK / 64, E_NUM);
    if (pre) {
        fc1_swiglu_kernel<1><<<g1, 256, 0, stream>>>(
            hs, hsb, fc1w, fc1wb, fc1b, counts, offsets, rows, ybuf);
        fc2_combine_kernel<1><<<g2, 256, 0, stream>>>(
            ybuf, fc2w, fc2wb, fc2b, counts, offsets, rows, rowgate, out);
    } else {
        fc1_swiglu_kernel<0><<<g1, 256, 0, stream>>>(
            hs, hsb, fc1w, fc1wb, fc1b, counts, offsets, rows, ybuf);
        fc2_combine_kernel<0><<<g2, 256, 0, stream>>>(
            ybuf, fc2w, fc2wb, fc2b, counts, offsets, rows, rowgate, out);
    }
}
